// GruDecoder_19344532702340
// MI455X (gfx1250) — compile-verified, matrix operand order checked by reading, not run on hardware
//
#include <hip/hip_runtime.h>
#include <math.h>

typedef __attribute__((ext_vector_type(16))) __bf16 v16bf;
typedef __attribute__((ext_vector_type(8)))  float  v8f;

#define GH 1024   // hidden
#define GO 768    // output / emb
#define GB 256    // batch
#define GT 256    // time steps
#define KK_IH 24  // GO/32  k-chunks for W_ih
#define KK_HH 32  // GH/32  k-chunks for W_hh
#define KK_FC 32  // GH/32  k-chunks for W_fc
#define LDS_BYTES 65536   // 2 panels x 32 k-chunks x 512 ushorts

__device__ __forceinline__ unsigned short f2bf(float f) {
    unsigned int u = __float_as_uint(f);
    u += 0x7FFFu + ((u >> 16) & 1u);          // round-to-nearest-even
    return (unsigned short)(u >> 16);
}

__device__ __forceinline__ float sigmoidf(float x) {
    return 1.0f / (1.0f + __expf(-x));
}

// ---------------------------------------------------------------------------
// Pack a row-major fp32 weight W[rows x K] (used as B = W^T, i.e. B[k][n] =
// W[n][k]) into bf16 WMMA B-fragments: fragment = (jt,kk) covering N=jt*16..+15,
// K=kk*32..+31, stored as 32 lanes x 16 bf16 contiguous (512 ushorts).
// Lane L: n = L%16, half = L/16; elem e: v=e/2,p=e%2,
//   k = (v>=4?16:0) + half*8 + (v&3)*2 + p
// ---------------------------------------------------------------------------
__global__ void pack_weights(const float* __restrict__ W, unsigned short* __restrict__ P,
                             int K, int NKK, int total) {
    int d = blockIdx.x * blockDim.x + threadIdx.x;
    if (d >= total) return;
    int e    = d & 15;
    int lane = (d >> 4) & 31;
    int frag = d >> 9;
    int kk   = frag % NKK;
    int jt   = frag / NKK;
    int n    = lane & 15;
    int hlf  = lane >> 4;
    int v = e >> 1, p = e & 1;
    int k = ((v >= 4) ? 16 : 0) + hlf * 8 + (v & 3) * 2 + p;
    P[d] = f2bf(W[(size_t)(jt * 16 + n) * K + kk * 32 + k]);
}

// ---------------------------------------------------------------------------
// Bulk-stage a 16 x (nkk*32) fp32 activation panel into LDS as nkk consecutive
// bf16 A-fragments (512 ushorts each; lane L: m = L%16, half = L/16; same k
// mapping as pack_weights). 128 threads; thread handles (r=tid/8, quad=tid%8)
// of every chunk kk.
// ---------------------------------------------------------------------------
__device__ __forceinline__ void stageFullA(const float* __restrict__ A, int ld,
                                           int row0, int nkk, unsigned short* lds) {
    const int r  = threadIdx.x >> 3;       // 0..15
    const int q  = threadIdx.x & 7;        // 0..7
    const int k4 = q << 2;                 // 0,4,...,28
    const float* rowp = A + (size_t)(row0 + r) * ld + k4;
    for (int kk = 0; kk < nkk; ++kk) {
        const float4 f4 = *(const float4*)(rowp + kk * 32);
        unsigned short* frag = lds + (kk << 9);
        float vals[4] = {f4.x, f4.y, f4.z, f4.w};
#pragma unroll
        for (int t = 0; t < 2; ++t) {
            int k    = k4 + t * 2;          // even
            int hlf  = (k >> 3) & 1;
            int lane = r + (hlf << 4);
            int v    = ((k >= 16) ? 4 : 0) + ((k & 7) >> 1);
            unsigned int pair = (unsigned int)f2bf(vals[t * 2]) |
                                ((unsigned int)f2bf(vals[t * 2 + 1]) << 16);
            *(unsigned int*)&frag[lane * 16 + (v << 1)] = pair;
        }
    }
}

// ---------------------------------------------------------------------------
// Fused GRU gate kernel: one wave = 32(batch) x 16(hidden) of h_new (two
// 16x16 m-tiles sharing each weight fragment -> halves L2 weight traffic).
// block = 128 (4 waves); grid = (B/32, H/64). 64KB dynamic LDS, reused
// across the two phases; K-loops are barrier-free.
// ---------------------------------------------------------------------------
__global__ __launch_bounds__(128)
void gru_gates(const float* __restrict__ xin,          // [B,GO]
               const float* __restrict__ hin,          // [B,GH]
               float* __restrict__ hout,               // [B,GH]
               const unsigned short* __restrict__ wihp,
               const unsigned short* __restrict__ whhp,
               const float* __restrict__ b_ih,
               const float* __restrict__ b_hh) {
    extern __shared__ unsigned short lds[];            // 64 KB
    const int wave = threadIdx.x >> 5;
    const int lane = threadIdx.x & 31;
    const int jt   = blockIdx.y * 4 + wave;            // 0..63
    const int mrow = blockIdx.x * 32;                  // first batch row

    v8f accR[2] = {{}, {}}, accZ[2] = {{}, {}}, accNI[2] = {{}, {}}, accNH[2] = {{}, {}};

    // ---- phase 1: gi = x * W_ih^T (K = GO) ------------------------------
    stageFullA(xin, GO, mrow,      KK_IH, lds);
    stageFullA(xin, GO, mrow + 16, KK_IH, lds + KK_IH * 512);
    __syncthreads();
    {
        const unsigned short* pr = wihp + (((size_t)(0 * 64 + jt) * KK_IH) << 9) + lane * 16;
        const unsigned short* pz = wihp + (((size_t)(1 * 64 + jt) * KK_IH) << 9) + lane * 16;
        const unsigned short* pn = wihp + (((size_t)(2 * 64 + jt) * KK_IH) << 9) + lane * 16;
#pragma unroll 2
        for (int kk = 0; kk < KK_IH; ++kk) {
            v16bf a0 = *(const v16bf*)&lds[(kk << 9) + lane * 16];
            v16bf a1 = *(const v16bf*)&lds[((KK_IH + kk) << 9) + lane * 16];
            v16bf br = *(const v16bf*)pr;  pr += 512;
            v16bf bz = *(const v16bf*)pz;  pz += 512;
            v16bf bn = *(const v16bf*)pn;  pn += 512;
            __builtin_prefetch((const void*)pr, 0, 1);
            __builtin_prefetch((const void*)pn, 0, 1);
            accR[0]  = __builtin_amdgcn_wmma_f32_16x16x32_bf16(false, a0, false, br, (short)0, accR[0],  false, false);
            accR[1]  = __builtin_amdgcn_wmma_f32_16x16x32_bf16(false, a1, false, br, (short)0, accR[1],  false, false);
            accZ[0]  = __builtin_amdgcn_wmma_f32_16x16x32_bf16(false, a0, false, bz, (short)0, accZ[0],  false, false);
            accZ[1]  = __builtin_amdgcn_wmma_f32_16x16x32_bf16(false, a1, false, bz, (short)0, accZ[1],  false, false);
            accNI[0] = __builtin_amdgcn_wmma_f32_16x16x32_bf16(false, a0, false, bn, (short)0, accNI[0], false, false);
            accNI[1] = __builtin_amdgcn_wmma_f32_16x16x32_bf16(false, a1, false, bn, (short)0, accNI[1], false, false);
        }
    }

    // ---- phase 2: gh = h * W_hh^T (K = GH) ------------------------------
    __syncthreads();                                   // done reading x panels
    stageFullA(hin, GH, mrow,      KK_HH, lds);
    stageFullA(hin, GH, mrow + 16, KK_HH, lds + KK_HH * 512);
    __syncthreads();
    {
        const unsigned short* pr = whhp + (((size_t)(0 * 64 + jt) * KK_HH) << 9) + lane * 16;
        const unsigned short* pz = whhp + (((size_t)(1 * 64 + jt) * KK_HH) << 9) + lane * 16;
        const unsigned short* pn = whhp + (((size_t)(2 * 64 + jt) * KK_HH) << 9) + lane * 16;
#pragma unroll 2
        for (int kk = 0; kk < KK_HH; ++kk) {
            v16bf a0 = *(const v16bf*)&lds[(kk << 9) + lane * 16];
            v16bf a1 = *(const v16bf*)&lds[((KK_HH + kk) << 9) + lane * 16];
            v16bf br = *(const v16bf*)pr;  pr += 512;
            v16bf bz = *(const v16bf*)pz;  pz += 512;
            v16bf bn = *(const v16bf*)pn;  pn += 512;
            __builtin_prefetch((const void*)pr, 0, 1);
            __builtin_prefetch((const void*)pn, 0, 1);
            accR[0]  = __builtin_amdgcn_wmma_f32_16x16x32_bf16(false, a0, false, br, (short)0, accR[0],  false, false);
            accR[1]  = __builtin_amdgcn_wmma_f32_16x16x32_bf16(false, a1, false, br, (short)0, accR[1],  false, false);
            accZ[0]  = __builtin_amdgcn_wmma_f32_16x16x32_bf16(false, a0, false, bz, (short)0, accZ[0],  false, false);
            accZ[1]  = __builtin_amdgcn_wmma_f32_16x16x32_bf16(false, a1, false, bz, (short)0, accZ[1],  false, false);
            accNH[0] = __builtin_amdgcn_wmma_f32_16x16x32_bf16(false, a0, false, bn, (short)0, accNH[0], false, false);
            accNH[1] = __builtin_amdgcn_wmma_f32_16x16x32_bf16(false, a1, false, bn, (short)0, accNH[1], false, false);
        }
    }

    // ---- epilogue: GRU nonlinearity. elem v -> (m = v + 8*half, n = lane%16)
    const int n   = lane & 15;
    const int hlf = lane >> 4;
    const int col = jt * 16 + n;
    const float br2 = b_ih[col]          + b_hh[col];
    const float bz2 = b_ih[GH + col]     + b_hh[GH + col];
    const float bni = b_ih[2 * GH + col];
    const float bnh = b_hh[2 * GH + col];
#pragma unroll
    for (int mi = 0; mi < 2; ++mi) {
#pragma unroll
        for (int v = 0; v < 8; ++v) {
            int m = mrow + mi * 16 + v + 8 * hlf;
            float r  = sigmoidf(accR[mi][v] + br2);
            float z  = sigmoidf(accZ[mi][v] + bz2);
            float nn = tanhf(accNI[mi][v] + bni + r * (accNH[mi][v] + bnh));
            float hp = hin[(size_t)m * GH + col];
            hout[(size_t)m * GH + col] = (1.0f - z) * nn + z * hp;
        }
    }
}

// ---------------------------------------------------------------------------
// fc feedback: y = h_new * W_fc^T + b_fc. One wave = 32x16 output tile.
// grid = (B/32, GO/64), block = 128, 64KB dynamic LDS.
// ---------------------------------------------------------------------------
__global__ __launch_bounds__(128)
void gru_fc(const float* __restrict__ hin,             // [B,GH]
            const unsigned short* __restrict__ wfcp,
            const float* __restrict__ b_fc,
            float* __restrict__ yout) {                // [B,GO]
    extern __shared__ unsigned short lds[];            // 64 KB
    const int wave = threadIdx.x >> 5;
    const int lane = threadIdx.x & 31;
    const int jt   = blockIdx.y * 4 + wave;            // 0..47
    const int mrow = blockIdx.x * 32;

    stageFullA(hin, GH, mrow,      KK_FC, lds);
    stageFullA(hin, GH, mrow + 16, KK_FC, lds + KK_FC * 512);
    __syncthreads();

    v8f acc[2] = {{}, {}};
    const unsigned short* pw = wfcp + (((size_t)jt * KK_FC) << 9) + lane * 16;
#pragma unroll 2
    for (int kk = 0; kk < KK_FC; ++kk) {
        v16bf a0 = *(const v16bf*)&lds[(kk << 9) + lane * 16];
        v16bf a1 = *(const v16bf*)&lds[((KK_FC + kk) << 9) + lane * 16];
        v16bf b  = *(const v16bf*)pw;  pw += 512;
        __builtin_prefetch((const void*)pw, 0, 1);
        acc[0] = __builtin_amdgcn_wmma_f32_16x16x32_bf16(false, a0, false, b, (short)0, acc[0], false, false);
        acc[1] = __builtin_amdgcn_wmma_f32_16x16x32_bf16(false, a1, false, b, (short)0, acc[1], false, false);
    }
    const int n   = lane & 15;
    const int hlf = lane >> 4;
    const int col = jt * 16 + n;
    const float bb = b_fc[col];
#pragma unroll
    for (int mi = 0; mi < 2; ++mi) {
#pragma unroll
        for (int v = 0; v < 8; ++v) {
            int m = mrow + mi * 16 + v + 8 * hlf;
            yout[(size_t)m * GO + col] = acc[mi][v] + bb;
        }
    }
}

// ---------------------------------------------------------------------------
extern "C" void kernel_launch(void* const* d_in, const int* in_sizes, int n_in,
                              void* d_out, int out_size, void* d_ws, size_t ws_size,
                              hipStream_t stream) {
    (void)in_sizes; (void)n_in; (void)out_size; (void)ws_size;
    const float* src    = (const float*)d_in[0];   // [2,B,GO], use src[0]
    const float* hidden = (const float*)d_in[2];   // [1,B,GH]
    const float* w_ih   = (const float*)d_in[3];   // [3H,GO]
    const float* w_hh   = (const float*)d_in[4];   // [3H,GH]
    const float* b_ih   = (const float*)d_in[5];
    const float* b_hh   = (const float*)d_in[6];
    const float* w_fc   = (const float*)d_in[7];   // [GO,GH]
    const float* b_fc   = (const float*)d_in[8];
    float* out = (float*)d_out;                    // [T,B,GO]

    // workspace layout
    char* ws = (char*)d_ws;
    const size_t sz_ih = (size_t)3 * GH * GO * sizeof(unsigned short);
    const size_t sz_hh = (size_t)3 * GH * GH * sizeof(unsigned short);
    const size_t sz_fc = (size_t)GO * GH * sizeof(unsigned short);
    unsigned short* wihp = (unsigned short*)ws;
    unsigned short* whhp = (unsigned short*)(ws + sz_ih);
    unsigned short* wfcp = (unsigned short*)(ws + sz_ih + sz_hh);
    float* h0 = (float*)(ws + sz_ih + sz_hh + sz_fc);
    float* h1 = h0 + (size_t)GB * GH;

    // pack weights to bf16 WMMA fragments (recomputed every call: deterministic)
    const int thr = 256;
    {
        int tot = 3 * GH * GO;
        pack_weights<<<(tot + thr - 1) / thr, thr, 0, stream>>>(w_ih, wihp, GO, KK_IH, tot);
    }
    {
        int tot = 3 * GH * GH;
        pack_weights<<<(tot + thr - 1) / thr, thr, 0, stream>>>(w_hh, whhp, GH, KK_HH, tot);
    }
    {
        int tot = GO * GH;
        pack_weights<<<(tot + thr - 1) / thr, thr, 0, stream>>>(w_fc, wfcp, GH, KK_FC, tot);
    }
    hipMemcpyAsync(h0, hidden, (size_t)GB * GH * sizeof(float),
                   hipMemcpyDeviceToDevice, stream);

    // sequential recurrence: 2 launches per step, stream-ordered (graph-safe)
    for (int t = 0; t < GT; ++t) {
        const float* xin  = (t == 0) ? src : (out + (size_t)(t - 1) * GB * GO);
        const float* hin  = (t & 1) ? h1 : h0;
        float*       hout = (t & 1) ? h0 : h1;
        gru_gates<<<dim3(GB / 32, GH / 64), 128, LDS_BYTES, stream>>>(
            xin, hin, hout, wihp, whhp, b_ih, b_hh);
        gru_fc<<<dim3(GB / 32, GO / 64), 128, LDS_BYTES, stream>>>(
            hout, wfcp, b_fc, out + (size_t)t * GB * GO);
    }
}
